// ConformerMultiHeadedSelfAttention_53206054863154
// MI455X (gfx1250) — matrix operand order repaired and checked by reading, not compiled, not run on hardware
//
#include <hip/hip_runtime.h>
#include <hip/hip_bf16.h>

#define BB  2
#define TT  2048
#define DD  512
#define HH  8
#define DHH 64
#define QTILES (TT/16)

typedef __bf16 bf16_t;
typedef __attribute__((ext_vector_type(16))) __bf16 v16bf;
typedef __attribute__((ext_vector_type(8)))  __bf16 v8bf;
typedef __attribute__((ext_vector_type(8)))  float  v8f;

__device__ __forceinline__ v16bf cat8(v8bf lo, v8bf hi) {
  return __builtin_shufflevector(lo, hi, 0,1,2,3,4,5,6,7,8,9,10,11,12,13,14,15);
}
// A operand (16x32 bf16): lane<16 holds K {0..7,16..23}, lane>=16 holds {8..15,24..31}.
// caller passes p = row_base + k0 + (laneHalf*8); chunks at +0..7 and +16..23.
__device__ __forceinline__ v16bf load_A_bf16(const bf16_t* p) {
  return cat8(*(const v8bf*)p, *(const v8bf*)(p + 16));
}
// B operand (32x16 bf16): lane<16 holds K {0..7,16..23}, lane>=16 holds {8..15,24..31}.
// caller passes p = row_base + k0 + (laneHalf*8); chunks at +0..7 and +16..23.
__device__ __forceinline__ v16bf load_B_bf16(const bf16_t* p) {
  return cat8(*(const v8bf*)p, *(const v8bf*)(p + 16));
}
// A operand built from f32 source (convert to bf16 on the fly)
__device__ __forceinline__ v16bf load_A_f32(const float* p) {
  v16bf a;
#pragma unroll
  for (int e = 0; e < 8; ++e) { a[e] = (bf16_t)p[e]; a[e + 8] = (bf16_t)p[e + 16]; }
  return a;
}
__device__ __forceinline__ v8f wmma_bf16(v16bf a, v16bf b, v8f c) {
  return __builtin_amdgcn_wmma_f32_16x16x32_bf16(false, a, false, b, (short)0, c, false, false);
}

// ---------------------------------------------------------------------------
// Weight transpose + f32 -> bf16 convert: WT[n*512+k] = (bf16)W[k*512+n]
// ---------------------------------------------------------------------------
__global__ __launch_bounds__(256) void wt_kernel(const float* __restrict__ W,
                                                 bf16_t* __restrict__ WT) {
  int idx = blockIdx.x * 256 + threadIdx.x;       // 0 .. 512*512-1
  int n = idx >> 9, k = idx & 511;
  WT[idx] = (bf16_t)W[k * 512 + n];
}

// ---------------------------------------------------------------------------
// Projection GEMM: X (Mtok x 512 f32) @ WT^T -> per-head bf16 tensors.
// mode 0: qu/qv (+bq, +u_bias / +v_bias)   mode 1: k    mode 2: vT   mode 3: p
// grid (Mtok/16, DD/256), block 128.  Each wave owns 64 output columns
// (4 WMMA accumulators) so one A load/convert feeds 4 WMMAs.
// ---------------------------------------------------------------------------
__global__ __launch_bounds__(128) void proj_kernel(
    const float* __restrict__ X, const bf16_t* __restrict__ WT, int mode,
    bf16_t* __restrict__ out0, bf16_t* __restrict__ out1,
    const float* __restrict__ bias, const float* __restrict__ ub,
    const float* __restrict__ vb) {
  const int m0   = blockIdx.x * 16;
  const int w    = threadIdx.x >> 5;
  const int lane = threadIdx.x & 31;
  const int half = lane >> 4;
  const int ln   = lane & 15;
  const int nb   = blockIdx.y * 256 + w * 64;     // wave's 64-col base

  const float*  xrow  = X  + (size_t)(m0 + ln) * DD;
  const bf16_t* wrow0 = WT + (size_t)(nb + ln) * DD;

  v8f acc[4] = {{}, {}, {}, {}};
  for (int k0 = 0; k0 < DD; k0 += 32) {
    v16bf A = load_A_f32(xrow + k0 + half * 8);
#pragma unroll
    for (int j = 0; j < 4; ++j) {
      v16bf B = load_B_bf16(wrow0 + (size_t)j * 16 * DD + k0 + half * 8);
      acc[j] = wmma_bf16(A, B, acc[j]);
    }
  }

#pragma unroll
  for (int j = 0; j < 4; ++j) {
    const int n  = nb + j * 16 + ln;
    const int hh = n >> 6, dh = n & 63;
#pragma unroll
    for (int r = 0; r < 8; ++r) {
      const int m = m0 + r + half * 8;
      if (mode == 3) {                                // p: [H, T, DH], m == t
        out0[((size_t)hh * TT + m) * DHH + dh] = (bf16_t)acc[j][r];
      } else {
        const int b = m >> 11, t = m & (TT - 1);
        if (mode == 0) {
          float base = acc[j][r] + bias[n];
          size_t o = (((size_t)(b * HH + hh)) * TT + t) * DHH + dh;
          out0[o] = (bf16_t)(base + ub[n]);           // q + u_bias
          out1[o] = (bf16_t)(base + vb[n]);           // q + v_bias
        } else if (mode == 1) {                       // k: [B,H,T,DH]
          out0[(((size_t)(b * HH + hh)) * TT + t) * DHH + dh] = (bf16_t)acc[j][r];
        } else {                                      // vT: [B,H,DH,T]
          out0[(((size_t)(b * HH + hh)) * DHH + dh) * TT + t] = (bf16_t)acc[j][r];
        }
      }
    }
  }
}

// ---------------------------------------------------------------------------
// Fused scores + rel_shift + mask + softmax + context.
// One workgroup = (b, h, 16-row q strip). Block 128 (4 waves).
// Dynamic LDS: 16 x 2048 f32 = 128 KB score row-block.
// ---------------------------------------------------------------------------
__global__ __launch_bounds__(128) void attn_kernel(
    const bf16_t* __restrict__ qu, const bf16_t* __restrict__ qv,
    const bf16_t* __restrict__ kb, const bf16_t* __restrict__ pb,
    const bf16_t* __restrict__ vT, const unsigned char* __restrict__ mask,
    float* __restrict__ attn_out, bf16_t* __restrict__ ctx) {
  extern __shared__ float S[];   // [16][TT]

  const int bid = blockIdx.x;
  const int qt  = bid % QTILES;
  const int h   = (bid / QTILES) % HH;
  const int b   = bid / (QTILES * HH);
  const int i0  = qt * 16;

  const int tid  = threadIdx.x;
  const int w    = tid >> 5;
  const int lane = tid & 31;
  const int half = lane >> 4;
  const int ln   = lane & 15;

  for (int idx = tid; idx < 16 * TT; idx += 128) S[idx] = 0.f;

  const bf16_t* qu_bh = qu + ((size_t)(b * HH + h)) * TT * DHH;
  const bf16_t* qv_bh = qv + ((size_t)(b * HH + h)) * TT * DHH;
  const bf16_t* k_bh  = kb + ((size_t)(b * HH + h)) * TT * DHH;
  const bf16_t* p_h   = pb + ((size_t)h) * TT * DHH;

  // Preload A operands (constant over the k-sweep).
  v16bf Aq0 = load_A_bf16(qu_bh + (i0 + ln) * DHH + half * 8);
  v16bf Aq1 = load_A_bf16(qu_bh + (i0 + ln) * DHH + 32 + half * 8);
  v16bf Av0 = load_A_bf16(qv_bh + (i0 + ln) * DHH + half * 8);
  v16bf Av1 = load_A_bf16(qv_bh + (i0 + ln) * DHH + 32 + half * 8);
  // Row-shifted qv (rows i0+1 .. i0+16) for the rel-shift "next row" term.
  v16bf Aw0 = load_A_bf16(qv_bh + (i0 + 1 + ln) * DHH + half * 8);
  v16bf Aw1 = load_A_bf16(qv_bh + (i0 + 1 + ln) * DHH + 32 + half * 8);
  __syncthreads();

  // ---- Sweep 1: positional scores, scattered through the rel-shift map ----
  // out(i,j) = pos(i, j-i+T-1) for j<=i ; 0 at j=i+1 ; pos(i+1, j-i-2) for j>=i+2
  for (int kt = w; kt < QTILES; kt += 4) {
    const bf16_t* prow = p_h + (size_t)(kt * 16 + ln) * DHH;
    v16bf B0 = load_B_bf16(prow + half * 8);
    v16bf B1 = load_B_bf16(prow + 32 + half * 8);
    v8f P  = {}; P  = wmma_bf16(Av0, B0, P);  P  = wmma_bf16(Av1, B1, P);
    v8f P2 = {}; P2 = wmma_bf16(Aw0, B0, P2); P2 = wmma_bf16(Aw1, B1, P2);
    const int c = kt * 16 + ln;
#pragma unroll
    for (int r = 0; r < 8; ++r) {
      const int M = r + half * 8;
      const int i = i0 + M;
      const int j1 = c + i - (TT - 1);
      if (j1 >= 0) S[M * TT + j1] = P[r];
      const int j2 = c + i + 2;
      if (j2 < TT) S[M * TT + j2] = P2[r];
    }
  }
  __syncthreads();

  // ---- Sweep 2: content scores, add shifted pos, scale, mask ----
  const float inv_sqrt_d = 0.044194173824159216f;   // 1/sqrt(512)
  for (int kt = w; kt < QTILES; kt += 4) {
    const bf16_t* krow = k_bh + (size_t)(kt * 16 + ln) * DHH;
    v16bf B0 = load_B_bf16(krow + half * 8);
    v16bf B1 = load_B_bf16(krow + 32 + half * 8);
    v8f C = {}; C = wmma_bf16(Aq0, B0, C); C = wmma_bf16(Aq1, B1, C);
    const int c = kt * 16 + ln;
    const bool mk = mask[b * TT + c] != 0;
#pragma unroll
    for (int r = 0; r < 8; ++r) {
      const int M = r + half * 8;
      float s = (C[r] + S[M * TT + c]) * inv_sqrt_d;
      if (mk) s = -1e9f;
      S[M * TT + c] = s;
    }
  }
  __syncthreads();

  // ---- Softmax (4 rows per wave), write attn to d_out, keep attn in LDS ----
  float* attn_bh = attn_out + (((size_t)(b * HH + h)) * TT + i0) * TT;
  for (int rr = 0; rr < 4; ++rr) {
    const int M = w * 4 + rr;
    float m = -3.4e38f;
    for (int c = lane; c < TT; c += 32) m = fmaxf(m, S[M * TT + c]);
#pragma unroll
    for (int off = 16; off; off >>= 1) m = fmaxf(m, __shfl_xor(m, off));
    float sum = 0.f;
    for (int c = lane; c < TT; c += 32) {
      float e = __expf(S[M * TT + c] - m);
      S[M * TT + c] = e;
      sum += e;
    }
#pragma unroll
    for (int off = 16; off; off >>= 1) sum += __shfl_xor(sum, off);
    const float inv = 1.f / sum;
    for (int c = lane; c < TT; c += 32) {
      float a = S[M * TT + c] * inv;
      S[M * TT + c] = a;
      attn_bh[(size_t)M * TT + c] = a;
    }
  }
  __syncthreads();

  // ---- Fused context GEMM: ctx_tile(16 x 64) = attn(16 x T) @ v(T x 64) ----
  const int n = w * 16 + ln;                              // col within head
  const bf16_t* vrow = vT + (((size_t)(b * HH + h)) * DHH + n) * TT;
  v8f acc = {};
  for (int j0 = 0; j0 < TT; j0 += 32) {
    const float* ab = &S[ln * TT + j0 + half * 8];
    v16bf A;
#pragma unroll
    for (int e = 0; e < 8; ++e) { A[e] = (bf16_t)ab[e]; A[e + 8] = (bf16_t)ab[e + 16]; }
    v16bf B = load_B_bf16(vrow + j0 + half * 8);
    acc = wmma_bf16(A, B, acc);
  }
  bf16_t* crow = ctx + ((size_t)(b * TT + i0)) * DD + h * DHH + n;
#pragma unroll
  for (int r = 0; r < 8; ++r) crow[(size_t)(r + half * 8) * DD] = (bf16_t)acc[r];
}

// ---------------------------------------------------------------------------
// Output GEMM: ctx (4096x512 bf16) @ WoT + bo -> f32 outputs.
// grid (4096/16, DD/256), block 128; each wave owns 64 cols (4 accumulators).
// ---------------------------------------------------------------------------
__global__ __launch_bounds__(128) void out_kernel(
    const bf16_t* __restrict__ ctx, const bf16_t* __restrict__ WoT,
    const float* __restrict__ bo, float* __restrict__ out) {
  const int m0   = blockIdx.x * 16;
  const int w    = threadIdx.x >> 5;
  const int lane = threadIdx.x & 31;
  const int half = lane >> 4;
  const int ln   = lane & 15;
  const int nb   = blockIdx.y * 256 + w * 64;

  const bf16_t* arow  = ctx + (size_t)(m0 + ln) * DD;
  const bf16_t* brow0 = WoT + (size_t)(nb + ln) * DD;

  v8f acc[4] = {{}, {}, {}, {}};
  for (int k0 = 0; k0 < DD; k0 += 32) {
    v16bf A = load_A_bf16(arow + k0 + half * 8);
#pragma unroll
    for (int j = 0; j < 4; ++j) {
      v16bf B = load_B_bf16(brow0 + (size_t)j * 16 * DD + k0 + half * 8);
      acc[j] = wmma_bf16(A, B, acc[j]);
    }
  }
#pragma unroll
  for (int j = 0; j < 4; ++j) {
    const int n  = nb + j * 16 + ln;
    const float bb = bo[n];
#pragma unroll
    for (int r = 0; r < 8; ++r)
      out[(size_t)(m0 + r + half * 8) * DD + n] = acc[j][r] + bb;
  }
}

// ---------------------------------------------------------------------------
extern "C" void kernel_launch(void* const* d_in, const int* in_sizes, int n_in,
                              void* d_out, int out_size, void* d_ws, size_t ws_size,
                              hipStream_t stream) {
  const float* query    = (const float*)d_in[0];
  const float* key      = (const float*)d_in[1];
  const float* value    = (const float*)d_in[2];
  const unsigned char* mask = (const unsigned char*)d_in[3];
  const float* encoding = (const float*)d_in[4];
  const float* Wq = (const float*)d_in[5];
  const float* bq = (const float*)d_in[6];
  const float* Wk = (const float*)d_in[7];
  const float* Wv = (const float*)d_in[8];
  const float* Wp = (const float*)d_in[9];
  const float* ub = (const float*)d_in[10];
  const float* vb = (const float*)d_in[11];
  const float* Wo = (const float*)d_in[12];
  const float* bo = (const float*)d_in[13];

  float* outputs  = (float*)d_out;                        // [B,T,D]
  float* attn_out = (float*)d_out + (size_t)BB * TT * DD; // [B,H,T,T]

  // Workspace carve-up (~26 MB)
  size_t off = 0;
  auto carve = [&](size_t bytes) {
    void* p = (char*)d_ws + off;
    off += (bytes + 255) & ~(size_t)255;
    return p;
  };
  const size_t wt_b  = (size_t)DD * DD * sizeof(bf16_t);
  const size_t act_b = (size_t)BB * HH * TT * DHH * sizeof(bf16_t);
  bf16_t* WTq = (bf16_t*)carve(wt_b);
  bf16_t* WTk = (bf16_t*)carve(wt_b);
  bf16_t* WTv = (bf16_t*)carve(wt_b);
  bf16_t* WTp = (bf16_t*)carve(wt_b);
  bf16_t* WTo = (bf16_t*)carve(wt_b);
  bf16_t* qu  = (bf16_t*)carve(act_b);
  bf16_t* qv  = (bf16_t*)carve(act_b + 256);   // +1 row pad for shifted overread
  bf16_t* kb  = (bf16_t*)carve(act_b);
  bf16_t* vT  = (bf16_t*)carve(act_b);
  bf16_t* pb  = (bf16_t*)carve((size_t)HH * TT * DHH * sizeof(bf16_t));
  bf16_t* ctx = (bf16_t*)carve((size_t)BB * TT * DD * sizeof(bf16_t));
  (void)ws_size; (void)n_in; (void)in_sizes; (void)out_size;

  // 1) Weight transpose/convert
  dim3 wtg(DD * DD / 256);
  wt_kernel<<<wtg, 256, 0, stream>>>(Wq, WTq);
  wt_kernel<<<wtg, 256, 0, stream>>>(Wk, WTk);
  wt_kernel<<<wtg, 256, 0, stream>>>(Wv, WTv);
  wt_kernel<<<wtg, 256, 0, stream>>>(Wp, WTp);
  wt_kernel<<<wtg, 256, 0, stream>>>(Wo, WTo);

  // 2) Projections (each wave computes 64 cols -> grid-y = DD/256 = 2)
  dim3 pg(BB * TT / 16, DD / 256);
  proj_kernel<<<pg, 128, 0, stream>>>(query, WTq, 0, qu, qv, bq, ub, vb);
  proj_kernel<<<pg, 128, 0, stream>>>(key,   WTk, 1, kb, nullptr, nullptr, nullptr, nullptr);
  proj_kernel<<<pg, 128, 0, stream>>>(value, WTv, 2, vT, nullptr, nullptr, nullptr, nullptr);
  dim3 eg(TT / 16, DD / 256);
  proj_kernel<<<eg, 128, 0, stream>>>(encoding, WTp, 3, pb, nullptr, nullptr, nullptr, nullptr);

  // 3) Fused attention (scores + rel_shift + softmax + context), 128 KB LDS
  attn_kernel<<<dim3(BB * HH * QTILES), 128, 16 * TT * sizeof(float), stream>>>(
      qu, qv, kb, pb, vT, mask, attn_out, ctx);

  // 4) Output projection
  out_kernel<<<dim3(BB * TT / 16, DD / 256), 128, 0, stream>>>(ctx, WTo, bo, outputs);
}
